// Transformer_72129680769662
// MI455X (gfx1250) — matrix operand order repaired and checked by reading, not compiled, not run on hardware
//
#include <hip/hip_runtime.h>
#include <hip/hip_bf16.h>
#include <math.h>

// ---------------------------------------------------------------------------
// Types for CDNA5 WMMA (wave32)
// ---------------------------------------------------------------------------
typedef __attribute__((ext_vector_type(16))) __bf16 bf16x16;
typedef __attribute__((ext_vector_type(8)))  __bf16 bf16x8;
typedef __attribute__((ext_vector_type(8)))  float  f32x8;
typedef __attribute__((ext_vector_type(4)))  unsigned int u32x4;
typedef __attribute__((ext_vector_type(8)))  int   i32x8;
typedef __attribute__((ext_vector_type(4)))  int   i32x4;

__device__ __forceinline__ __bf16 to_bf16(float f) { return (__bf16)f; }

__device__ __forceinline__ f32x8 wmma_bf16(bf16x16 a, bf16x16 b, f32x8 c) {
  // D = A(16x32 bf16) * B(32x16 bf16) + C(16x16 f32)
  return __builtin_amdgcn_wmma_f32_16x16x32_bf16(
      /*neg_a=*/false, a, /*neg_b=*/false, b,
      /*c_mod=*/(short)0, c, /*reuse_a=*/false, /*reuse_b=*/false);
}

// ---------------------------------------------------------------------------
// Tensor Data Mover: async 2D bf16 tile load (global -> LDS), TENSORcnt.
// Builds a D# (groups 0/1; remaining groups zero for a 2D tensor) per
// CDNA5 ISA and issues TENSOR_LOAD_TO_LDS.
// This toolchain exposes the 6-arg builtin: (g0, g1, g2, g3, extra, cpol).
// ---------------------------------------------------------------------------
__device__ __forceinline__ void tdm_load_tile_bf16(
    unsigned lds_addr,            // byte offset in LDS
    const void* gptr,             // global address of tile start
    unsigned tile_cols,           // elements per row (tile_dim0)
    unsigned tile_rows,           // rows (tile_dim1)
    unsigned tensor_cols,         // tensor_dim0 (elements)
    unsigned tensor_rows,         // tensor_dim1 (rows)
    unsigned row_stride_elems) {  // tensor_dim0_stride (elements)
  unsigned long long ga = (unsigned long long)gptr;
  u32x4 g0;
  g0[0] = 1u;                                            // count=1, user mode
  g0[1] = lds_addr;                                      // lds_addr [63:32]
  g0[2] = (unsigned)(ga & 0xFFFFFFFFu);                  // global_addr lo
  g0[3] = (unsigned)((ga >> 32) & 0x01FFFFFFu) | (2u << 30);  // addr hi | type=2
  i32x8 g1;
  g1[0] = (int)(1u << 16);                               // data_size=1 (2B), mask=0
  g1[1] = (int)((tensor_cols & 0xFFFFu) << 16);          // tensor_dim0[15:0]
  g1[2] = (int)((tensor_cols >> 16) |
                ((tensor_rows & 0xFFFFu) << 16));        // dim0 hi | dim1 lo
  g1[3] = (int)((tensor_rows >> 16) |
                ((tile_cols & 0xFFFFu) << 16));          // dim1 hi | tile_dim0
  g1[4] = (int)(tile_rows & 0xFFFFu);                    // tile_dim1 | tile_dim2=0
  g1[5] = (int)row_stride_elems;                         // dim0_stride[31:0]
  g1[6] = 0;                                             // stride hi | dim1_stride lo
  g1[7] = 0;                                             // dim1_stride hi
  i32x4 z4 = {0, 0, 0, 0};
  i32x8 z8 = {0, 0, 0, 0, 0, 0, 0, 0};
  __builtin_amdgcn_tensor_load_to_lds(g0, g1, z4, z4, z8, 0);
}

__device__ __forceinline__ unsigned lds_off(const void* p) {
  // Generic shared pointer -> 32-bit LDS byte address (addr[31:0] per ISA).
  return (unsigned)(size_t)p;
}

#define DIMM      1024
#define HEADS     16
#define DIM_HEAD  64
#define SEQ       2048
#define BATCH     2
#define MTOT      (BATCH * SEQ)       // 4096 flattened rows

// ---------------------------------------------------------------------------
// Utility kernels
// ---------------------------------------------------------------------------
__global__ __launch_bounds__(256) void copy_f32(const float* __restrict__ in,
                                                float* out, size_t n4) {
  size_t i = ((size_t)blockIdx.x * 256 + threadIdx.x) * 4;
  if (i < n4) *(float4*)(out + i) = *(const float4*)(in + i);
}

// W (K x N, f32, row-major) -> Wt (N x K, bf16, row-major)
__global__ __launch_bounds__(256) void cast_transpose(const float* __restrict__ W,
                                                      __bf16* __restrict__ Wt,
                                                      int K, int N) {
  size_t idx = (size_t)blockIdx.x * 256 + threadIdx.x;
  if (idx >= (size_t)K * N) return;
  int k = (int)(idx / N), n = (int)(idx % N);
  Wt[(size_t)n * K + k] = to_bf16(W[idx]);
}

// rel-pos bias table: tab[h][d] = rel_emb[bucket(d)][h] * sqrt(64)
__global__ __launch_bounds__(256) void bias_table(const float* __restrict__ rel_emb,
                                                  float* __restrict__ tab) {
  int idx = blockIdx.x * 256 + threadIdx.x;
  if (idx >= HEADS * SEQ) return;
  int h = idx / SEQ, d = idx % SEQ;
  int bucket;
  if (d < 16) {
    bucket = d;
  } else {
    int v = 16 + (int)(logf((float)d * (1.0f / 16.0f)) * (16.0f / logf(6.0f)));
    bucket = v < 31 ? v : 31;
  }
  tab[idx] = rel_emb[bucket * HEADS + h] * 8.0f;   // BIAS_SCALE = sqrt(DIM_HEAD)
}

// ---------------------------------------------------------------------------
// Fused LayerNorm (fp32 in) -> bf16 out.  One block per row of 1024.
// ---------------------------------------------------------------------------
__global__ __launch_bounds__(256) void layernorm_bf16(const float* __restrict__ x,
                                                      const float* __restrict__ w,
                                                      const float* __restrict__ b,
                                                      __bf16* __restrict__ out) {
  int row = blockIdx.x;
  const float* xr = x + (size_t)row * DIMM;
  int tid = threadIdx.x;
  float4 v = *(const float4*)(xr + tid * 4);
  float s  = v.x + v.y + v.z + v.w;
  float sq = v.x * v.x + v.y * v.y + v.z * v.z + v.w * v.w;
  #pragma unroll
  for (int m = 1; m < 32; m <<= 1) {
    s  += __shfl_xor(s, m, 32);
    sq += __shfl_xor(sq, m, 32);
  }
  __shared__ float ws[8], wsq[8];
  __shared__ float mu_s, rstd_s;
  int lane = tid & 31, wv = tid >> 5;
  if (lane == 0) { ws[wv] = s; wsq[wv] = sq; }
  __syncthreads();
  if (tid == 0) {
    float S = 0.f, SQ = 0.f;
    #pragma unroll
    for (int i = 0; i < 8; i++) { S += ws[i]; SQ += wsq[i]; }
    float mu  = S * (1.0f / DIMM);
    float var = SQ * (1.0f / DIMM) - mu * mu;
    mu_s = mu; rstd_s = rsqrtf(var + 1e-5f);
  }
  __syncthreads();
  float mu = mu_s, rstd = rstd_s;
  __bf16* orow = out + (size_t)row * DIMM;
  int c = tid * 4;
  orow[c + 0] = to_bf16((v.x - mu) * rstd * w[c + 0] + b[c + 0]);
  orow[c + 1] = to_bf16((v.y - mu) * rstd * w[c + 1] + b[c + 1]);
  orow[c + 2] = to_bf16((v.z - mu) * rstd * w[c + 2] + b[c + 2]);
  orow[c + 3] = to_bf16((v.w - mu) * rstd * w[c + 3] + b[c + 3]);
}

// ---------------------------------------------------------------------------
// bf16 WMMA GEMM, TDM-staged and double-buffered.
// Tile 128x64, BK=32, 8 waves: wave w owns rows [w*16, w*16+16) and all 64
// cols -> 4 accumulators, 4 WMMA per k-step off one A-fragment.
// Wave 0 drives the Tensor Data Mover: while buffer `buf` is being consumed,
// the next K-slice streams into buffer `buf^1` (TENSORcnt + barriers).
// ---------------------------------------------------------------------------
#define GBM 128
#define GBN 64
#define GBK 32

__global__ __launch_bounds__(256) void gemm_bf16(
    const __bf16* __restrict__ A, const __bf16* __restrict__ Bt,
    int M, int N, int K,
    __bf16* outBf, float* outF,
    const float* __restrict__ bias, const float* resid,
    float alpha, int gelu) {
  __shared__ alignas(32) __bf16 sA[2][GBM * GBK];
  __shared__ alignas(32) __bf16 sB[2][GBN * GBK];

  int tid = threadIdx.x;
  int lane = tid & 31, wave = tid >> 5;
  int mbase = blockIdx.y * GBM, nbase = blockIdx.x * GBN;
  int l16 = lane & 15, khalf = lane >> 4;
  bool issuer = (wave == 0);

  const __bf16* Abase  = A  + (size_t)mbase * K;
  const __bf16* Btbase = Bt + (size_t)nbase * K;

  f32x8 acc[4] = {{}, {}, {}, {}};

  if (issuer) {
    tdm_load_tile_bf16(lds_off(&sA[0][0]), Abase,  GBK, GBM, K, M, K);
    tdm_load_tile_bf16(lds_off(&sB[0][0]), Btbase, GBK, GBN, K, N, K);
  }

  int buf = 0;
  for (int k0 = 0; k0 < K; k0 += GBK, buf ^= 1) {
    if (issuer) __builtin_amdgcn_s_wait_tensorcnt(0);
    __syncthreads();                      // publish tile `buf` to all waves
    if (issuer && (k0 + GBK) < K) {       // prefetch next slice into buf^1
      tdm_load_tile_bf16(lds_off(&sA[buf ^ 1][0]), Abase  + k0 + GBK,
                         GBK, GBM, K, M, K);
      tdm_load_tile_bf16(lds_off(&sB[buf ^ 1][0]), Btbase + k0 + GBK,
                         GBK, GBN, K, N, K);
    }

    // A fragment (16x32): lane row = wave*16 + l16, K split by lane-half:
    // elems e -> K = 16*khalf + e (contig)
    const __bf16* ap = &sA[buf][(wave * 16 + l16) * GBK + khalf * 16];
    bf16x8 alo = *(const bf16x8*)ap;
    bf16x8 ahi = *(const bf16x8*)(ap + 8);
    bf16x16 afrag;
    #pragma unroll
    for (int e = 0; e < 8; e++) { afrag[e] = alo[e]; afrag[e + 8] = ahi[e]; }

    // B fragments (32x16): lane col = n, elems e -> K = 16*khalf + e (contig)
    #pragma unroll
    for (int t = 0; t < 4; t++) {
      bf16x16 bfrag =
          *(const bf16x16*)(&sB[buf][(t * 16 + l16) * GBK + khalf * 16]);
      acc[t] = wmma_bf16(afrag, bfrag, acc[t]);
    }
    __syncthreads();                      // done reading tile `buf`
  }

  // Epilogue.  C layout: VGPR r, lane -> M = r + 8*(lane/16), N = lane%16.
  int mrow0 = mbase + wave * 16 + khalf * 8;
  #pragma unroll
  for (int t = 0; t < 4; t++) {
    int n = nbase + t * 16 + l16;
    float bv = bias ? bias[n] : 0.0f;
    #pragma unroll
    for (int rr = 0; rr < 8; rr++) {
      int m = mrow0 + rr;
      float v = acc[t][rr] * alpha + bv;
      if (gelu) v = 0.5f * v * (1.0f + erff(v * 0.70710678f));
      size_t off = (size_t)m * N + n;
      if (outF)  outF[off]  = v + (resid ? resid[off] : 0.0f);
      if (outBf) outBf[off] = to_bf16(v);
    }
  }
}

// ---------------------------------------------------------------------------
// Flash attention (multi-query): one workgroup per (b, h, 128 query rows).
// Wave w owns 16 query rows. K/V streamed as 32-key blocks through LDS.
// S = Q K^T via WMMA, online softmax (shfl row-reduce over lane halves),
// P re-laid out through wave-private LDS into an A-fragment, O += P V.
// q is pre-scaled by DIM_HEAD^-0.5 in the Q projection epilogue.
// ---------------------------------------------------------------------------
#define AQB 128
#define AKB 32

__global__ __launch_bounds__(256) void attention(
    const __bf16* __restrict__ q,    // [MTOT, 1024]  (h*64+d)
    const __bf16* __restrict__ kv,   // [MTOT, 128]   (K | V)
    const float* __restrict__ biasTab,  // [HEADS, SEQ]
    __bf16* __restrict__ out) {      // [MTOT, 1024]
  __shared__ alignas(32) __bf16 sK[AKB * DIM_HEAD];   // [j][d]
  __shared__ alignas(32) __bf16 sVt[DIM_HEAD * AKB];  // [d][j]
  __shared__ alignas(32) __bf16 sP[8 * 16 * AKB];     // per-wave P scratch

  int tid = threadIdx.x, lane = tid & 31, wave = tid >> 5;
  int l16 = lane & 15, khalf = lane >> 4;
  int bh = blockIdx.y;
  int b = bh >> 4, h = bh & 15;
  int qb = blockIdx.x;
  int qrow0 = qb * AQB + wave * 16;

  // Q A-fragments for two 32-wide k-steps over head dim
  const __bf16* qrow = q + ((size_t)(b * SEQ + qrow0 + l16)) * DIMM + h * DIM_HEAD;
  bf16x16 qf[2];
  #pragma unroll
  for (int s = 0; s < 2; s++) {
    bf16x8 lo = *(const bf16x8*)(qrow + s * 32 + khalf * 16);
    bf16x8 hi = *(const bf16x8*)(qrow + s * 32 + khalf * 16 + 8);
    #pragma unroll
    for (int e = 0; e < 8; e++) { qf[s][e] = lo[e]; qf[s][e + 8] = hi[e]; }
  }

  f32x8 o0 = {}, o1 = {}, o2 = {}, o3 = {};
  float mrow[8], lrow[8];
  #pragma unroll
  for (int i = 0; i < 8; i++) { mrow[i] = -3.4e38f; lrow[i] = 0.0f; }

  const __bf16* kvb = kv + (size_t)b * SEQ * 128;
  int kmax = (qb + 1) * AQB;

  for (int kb = 0; kb < kmax; kb += AKB) {
    // ---- stage K (row-major) and V (transposed) into LDS ----
    {
      int j = tid >> 3, c8 = (tid & 7) * 8;
      bf16x8 kx = *(const bf16x8*)(kvb + (size_t)(kb + j) * 128 + c8);
      *(bf16x8*)(sK + j * DIM_HEAD + c8) = kx;
      bf16x8 vx = *(const bf16x8*)(kvb + (size_t)(kb + j) * 128 + 64 + c8);
      #pragma unroll
      for (int e = 0; e < 8; e++) sVt[(c8 + e) * AKB + j] = vx[e];
    }
    __syncthreads();

    if (kb <= qrow0 + 15) {   // causal: this wave touches this key block
      // ---- S = Q K^T  (two 16-key tiles) ----
      f32x8 s0 = {}, s1 = {};
      #pragma unroll
      for (int s = 0; s < 2; s++) {
        bf16x16 bk0 = *(const bf16x16*)(sK + l16 * DIM_HEAD + s * 32 + khalf * 16);
        bf16x16 bk1 = *(const bf16x16*)(sK + (l16 + 16) * DIM_HEAD + s * 32 + khalf * 16);
        s0 = wmma_bf16(qf[s], bk0, s0);
        s1 = wmma_bf16(qf[s], bk1, s1);
      }
      // ---- bias + causal mask ----
      int i0 = qrow0 + khalf * 8;
      int j0 = kb + l16, j1 = kb + 16 + l16;
      #pragma unroll
      for (int rr = 0; rr < 8; rr++) {
        int i = i0 + rr;
        s0[rr] = (j0 <= i) ? (s0[rr] + biasTab[h * SEQ + (i - j0)]) : -3.4e38f;
        s1[rr] = (j1 <= i) ? (s1[rr] + biasTab[h * SEQ + (i - j1)]) : -3.4e38f;
      }
      // ---- online softmax (row-reduce across the 16 lanes of each half) ----
      #pragma unroll
      for (int rr = 0; rr < 8; rr++) {
        float mx = fmaxf(s0[rr], s1[rr]);
        #pragma unroll
        for (int msk = 1; msk < 16; msk <<= 1) mx = fmaxf(mx, __shfl_xor(mx, msk, 32));
        float mnew = fmaxf(mrow[rr], mx);
        float scale = expf(mrow[rr] - mnew);
        float p0 = expf(s0[rr] - mnew), p1 = expf(s1[rr] - mnew);
        s0[rr] = p0; s1[rr] = p1;
        float ps = p0 + p1;
        #pragma unroll
        for (int msk = 1; msk < 16; msk <<= 1) ps += __shfl_xor(ps, msk, 32);
        lrow[rr] = lrow[rr] * scale + ps;
        mrow[rr] = mnew;
        o0[rr] *= scale; o1[rr] *= scale; o2[rr] *= scale; o3[rr] *= scale;
      }
      // ---- P (C-layout) -> wave-private LDS -> A-fragment ----
      __bf16* pscr = sP + wave * 16 * AKB;
      #pragma unroll
      for (int rr = 0; rr < 8; rr++) {
        int m = rr + khalf * 8;
        pscr[m * AKB + l16]      = to_bf16(s0[rr]);
        pscr[m * AKB + 16 + l16] = to_bf16(s1[rr]);
      }
      bf16x16 pf;
      {
        const __bf16* pp = pscr + l16 * AKB + khalf * 16;
        bf16x8 lo = *(const bf16x8*)pp;
        bf16x8 hi = *(const bf16x8*)(pp + 8);
        #pragma unroll
        for (int e = 0; e < 8; e++) { pf[e] = lo[e]; pf[e + 8] = hi[e]; }
      }
      // ---- O += P V  (V^T staged, contiguous B-fragments) ----
      bf16x16 bv0 = *(const bf16x16*)(sVt + (l16 +  0) * AKB + khalf * 16);
      bf16x16 bv1 = *(const bf16x16*)(sVt + (l16 + 16) * AKB + khalf * 16);
      bf16x16 bv2 = *(const bf16x16*)(sVt + (l16 + 32) * AKB + khalf * 16);
      bf16x16 bv3 = *(const bf16x16*)(sVt + (l16 + 48) * AKB + khalf * 16);
      o0 = wmma_bf16(pf, bv0, o0);
      o1 = wmma_bf16(pf, bv1, o1);
      o2 = wmma_bf16(pf, bv2, o2);
      o3 = wmma_bf16(pf, bv3, o3);
    }
    __syncthreads();
  }

  // ---- normalize and store bf16 (feeds O-projection GEMM) ----
  #pragma unroll
  for (int rr = 0; rr < 8; rr++) {
    int i = qrow0 + rr + khalf * 8;
    float inv = 1.0f / lrow[rr];
    size_t base = ((size_t)(b * SEQ + i)) * DIMM + h * DIM_HEAD;
    out[base +      l16] = to_bf16(o0[rr] * inv);
    out[base + 16 + l16] = to_bf16(o1[rr] * inv);
    out[base + 32 + l16] = to_bf16(o2[rr] * inv);
    out[base + 48 + l16] = to_bf16(o3[rr] * inv);
  }
}

// ---------------------------------------------------------------------------
// Host side
// ---------------------------------------------------------------------------
struct LayerIn {
  const float *b1, *b2, *bo, *ln1_b, *ln1_w, *ln2_b, *ln2_w, *w1, *w2, *wkv, *wo, *wq;
};

extern "C" void kernel_launch(void* const* d_in, const int* in_sizes, int n_in,
                              void* d_out, int out_size, void* d_ws, size_t ws_size,
                              hipStream_t stream) {
  (void)in_sizes; (void)n_in; (void)out_size; (void)ws_size;
  // Input order: x, 4 layers x 12 leaves (sorted keys), rel_emb, mask
  const float* x_in = (const float*)d_in[0];
  LayerIn L[4];
  for (int i = 0; i < 4; i++) {
    const int base = 1 + i * 12;
    L[i].b1    = (const float*)d_in[base + 0];
    L[i].b2    = (const float*)d_in[base + 1];
    L[i].bo    = (const float*)d_in[base + 2];
    L[i].ln1_b = (const float*)d_in[base + 3];
    L[i].ln1_w = (const float*)d_in[base + 4];
    L[i].ln2_b = (const float*)d_in[base + 5];
    L[i].ln2_w = (const float*)d_in[base + 6];
    L[i].w1    = (const float*)d_in[base + 7];
    L[i].w2    = (const float*)d_in[base + 8];
    L[i].wkv   = (const float*)d_in[base + 9];
    L[i].wo    = (const float*)d_in[base + 10];
    L[i].wq    = (const float*)d_in[base + 11];
  }
  const float* rel_emb = (const float*)d_in[49];
  float* x = (float*)d_out;   // residual stream lives in d_out

  // Workspace layout
  char* ws = (char*)d_ws;
  size_t off = 0;
  auto wsAlloc = [&](size_t bytes) {
    char* p = ws + off;
    off += (bytes + 255) & ~(size_t)255;
    return p;
  };
  __bf16* hbuf   = (__bf16*)wsAlloc((size_t)MTOT * DIMM * 2);
  __bf16* qbuf   = (__bf16*)wsAlloc((size_t)MTOT * DIMM * 2);
  __bf16* kvbuf  = (__bf16*)wsAlloc((size_t)MTOT * 128 * 2);
  __bf16* abuf   = (__bf16*)wsAlloc((size_t)MTOT * DIMM * 2);
  __bf16* ffbuf  = (__bf16*)wsAlloc((size_t)MTOT * 4096 * 2);
  __bf16* wqT    = (__bf16*)wsAlloc((size_t)1024 * 1024 * 2);
  __bf16* wkvT   = (__bf16*)wsAlloc((size_t)128 * 1024 * 2);
  __bf16* woT    = (__bf16*)wsAlloc((size_t)1024 * 1024 * 2);
  __bf16* w1T    = (__bf16*)wsAlloc((size_t)4096 * 1024 * 2);
  __bf16* w2T    = (__bf16*)wsAlloc((size_t)1024 * 4096 * 2);
  float*  btab   = (float*) wsAlloc((size_t)HEADS * SEQ * 4);

  // x <- input; bias table
  copy_f32<<<(MTOT * DIMM / 4 + 255) / 256, 256, 0, stream>>>(x_in, x, (size_t)MTOT * DIMM);
  bias_table<<<(HEADS * SEQ + 255) / 256, 256, 0, stream>>>(rel_emb, btab);

  auto gemm = [&](const __bf16* A, const __bf16* Bt, int M, int N, int K,
                  __bf16* outBf, float* outF, const float* bias,
                  const float* resid, float alpha, int gelu) {
    dim3 grid(N / GBN, M / GBM);
    gemm_bf16<<<grid, 256, 0, stream>>>(A, Bt, M, N, K, outBf, outF, bias,
                                        resid, alpha, gelu);
  };
  auto castT = [&](const float* W, __bf16* Wt, int K, int N) {
    size_t n = (size_t)K * N;
    cast_transpose<<<(unsigned)((n + 255) / 256), 256, 0, stream>>>(W, Wt, K, N);
  };

  const float qscale = 0.125f;  // DIM_HEAD^-0.5

  for (int i = 0; i < 4; i++) {
    // weights -> bf16 transposed
    castT(L[i].wq,  wqT,  1024, 1024);
    castT(L[i].wkv, wkvT, 1024, 128);
    castT(L[i].wo,  woT,  1024, 1024);
    castT(L[i].w1,  w1T,  1024, 4096);
    castT(L[i].w2,  w2T,  4096, 1024);

    // LN1 -> h (bf16)
    layernorm_bf16<<<MTOT, 256, 0, stream>>>(x, L[i].ln1_w, L[i].ln1_b, hbuf);

    // Q (pre-scaled) and KV projections
    gemm(hbuf, wqT,  MTOT, 1024, 1024, qbuf,  nullptr, nullptr, nullptr, qscale, 0);
    gemm(hbuf, wkvT, MTOT, 128,  1024, kvbuf, nullptr, nullptr, nullptr, 1.0f,   0);

    // attention
    {
      dim3 grid(SEQ / AQB, BATCH * HEADS);
      attention<<<grid, 256, 0, stream>>>(qbuf, kvbuf, btab, abuf);
    }

    // O projection + bias + residual (fp32 stream in d_out)
    gemm(abuf, woT, MTOT, 1024, 1024, nullptr, x, L[i].bo, x, 1.0f, 0);

    // LN2 -> h (bf16)
    layernorm_bf16<<<MTOT, 256, 0, stream>>>(x, L[i].ln2_w, L[i].ln2_b, hbuf);

    // FF1 + GELU (bf16 out), FF2 + bias + residual
    gemm(hbuf,  w1T, MTOT, 4096, 1024, ffbuf, nullptr, L[i].b1, nullptr, 1.0f, 1);
    gemm(ffbuf, w2T, MTOT, 1024, 4096, nullptr, x, L[i].b2, x, 1.0f, 0);
  }
}
